// VectorQuantizer_29961691857520
// MI455X (gfx1250) — compile-verified, matrix operand order checked by reading, not run on hardware
//
#include <hip/hip_runtime.h>

#define NUM_CODES 1024
#define DIM       64
#define NROWS     (64 * 8192)          // 524288 rows
#define TILES     (NUM_CODES / 16)     // 64 code tiles
#define ARR_BYTES (TILES * 2 * 32 * 16 * 2)  // 131072 bytes of swizzled bf16 codebook
#define SMEM_BYTES (ARR_BYTES + NUM_CODES * 4) // + 4KB norms = 135168

typedef __attribute__((ext_vector_type(16))) __bf16 v16bf;
typedef __attribute__((ext_vector_type(8)))  float  v8f;
typedef __attribute__((ext_vector_type(2)))  float  f32x2;
typedef __attribute__((ext_vector_type(4)))  float  f32x4;

__device__ __forceinline__ unsigned short f2bf(float f) {
    unsigned int u = __float_as_uint(f);
    unsigned int r = u + 0x7FFFu + ((u >> 16) & 1u);   // round-to-nearest-even
    return (unsigned short)(r >> 16);
}

// ---------------------------------------------------------------------------
// Kernel 1: codebook norms + swizzle codebook into WMMA B-fragment layout.
// arranged[((t*2+chunk)*32 + lane)*16 + e], e=2p/2p+1 pairs along K.
// ---------------------------------------------------------------------------
__global__ void vq_prep(const float* __restrict__ cb,
                        unsigned short* __restrict__ arranged,
                        float* __restrict__ norms) {
    int i = blockIdx.x * blockDim.x + threadIdx.x;
    if (i < TILES * 2 * 32 * 16) {
        int e     = i & 15;
        int lane  = (i >> 4) & 31;
        int chunk = (i >> 9) & 1;
        int t     = i >> 10;
        int p     = e >> 1;
        int n     = t * 16 + (lane & 15);                       // code (B column)
        int k     = chunk * 32 + ((p < 4) ? 0 : 16)
                  + ((lane >> 4) << 3) + ((p & 3) << 1) + (e & 1); // K index
        arranged[i] = f2bf(cb[n * DIM + k]);
    }
    if (i < NUM_CODES) {
        float s = 0.0f;
        #pragma unroll 8
        for (int d = 0; d < DIM; ++d) { float v = cb[i * DIM + d]; s += v * v; }
        norms[i] = s;
    }
}

// ---------------------------------------------------------------------------
// Kernel 2: main distance + argmin. 8 waves/block, 16 rows/wave.
// dist = ||e||^2 - 2 x.e  (||x||^2 dropped: constant per row).
// A operand pre-scaled by -2; C initialized with ||e||^2, so the WMMA result
// IS the distance. Column index is embedded in the low 10 mantissa bits so
// argmin tracking is a single v_and_or_b32 + v_min_num_f32 per register.
// ---------------------------------------------------------------------------
__global__ __launch_bounds__(256) void vq_main(const float* __restrict__ x,
                                               const unsigned short* __restrict__ arranged,
                                               const float* __restrict__ norms,
                                               float* __restrict__ idxOut) {
    extern __shared__ unsigned char smem[];
    uint4* sArr  = (uint4*)smem;                       // 8192 uint4
    float* sNorm = (float*)(smem + ARR_BYTES);         // 1024 floats

    int tid = threadIdx.x;
    // Stage swizzled codebook + norms into LDS once per block.
    const uint4* gArr = (const uint4*)arranged;
    for (int i = tid; i < ARR_BYTES / 16; i += 256) sArr[i] = gArr[i];
    if (tid < 256) ((uint4*)sNorm)[tid] = ((const uint4*)norms)[tid];
    __syncthreads();

    int wave = tid >> 5;
    int lane = tid & 31;
    int m    = lane & 15;        // A row / C column within tile
    int half = lane >> 4;        // K-half selector for A/B, M-half for C
    long rowbase = (long)blockIdx.x * 128 + wave * 16;
    const float* rowp = x + (rowbase + m) * DIM;

    // A fragments for K=0..31 and K=32..63, converted to bf16 scaled by -2.
    union { v16bf v; unsigned int u[8]; } a0, a1;
    #pragma unroll
    for (int p = 0; p < 8; ++p) {
        int k = ((p < 4) ? 0 : 16) + half * 8 + ((p & 3) << 1);
        f32x2 f0 = __builtin_nontemporal_load((const f32x2*)(rowp + k));
        f32x2 f1 = __builtin_nontemporal_load((const f32x2*)(rowp + 32 + k));
        a0.u[p] = (unsigned int)f2bf(-2.0f * f0.x) | ((unsigned int)f2bf(-2.0f * f0.y) << 16);
        a1.u[p] = (unsigned int)f2bf(-2.0f * f1.x) | ((unsigned int)f2bf(-2.0f * f1.y) << 16);
    }

    float best[8];
    #pragma unroll
    for (int j = 0; j < 8; ++j) best[j] = 3.4e38f;

    #pragma unroll 4
    for (int t = 0; t < TILES; ++t) {
        union { v16bf v; uint4 q[2]; } b0, b1;
        int f0i = ((t * 2 + 0) * 32 + lane) * 2;
        int f1i = ((t * 2 + 1) * 32 + lane) * 2;
        b0.q[0] = sArr[f0i];     b0.q[1] = sArr[f0i + 1];
        b1.q[0] = sArr[f1i];     b1.q[1] = sArr[f1i + 1];

        int   col = t * 16 + m;
        float cn  = sNorm[col];
        v8f c;
        #pragma unroll
        for (int j = 0; j < 8; ++j) c[j] = cn;          // C init = ||e||^2

        c = __builtin_amdgcn_wmma_f32_16x16x32_bf16(false, a0.v, false, b0.v,
                                                    (short)0, c, false, false);
        c = __builtin_amdgcn_wmma_f32_16x16x32_bf16(false, a1.v, false, b1.v,
                                                    (short)0, c, false, false);

        #pragma unroll
        for (int j = 0; j < 8; ++j) {
            // key = dist with column index embedded in the low 10 mantissa bits
            float key = __uint_as_float((__float_as_uint(c[j]) & 0xFFFFFC00u)
                                        | (unsigned)col);
            best[j] = fminf(best[j], key);
        }
    }

    // Reduce min across the 16 columns (lanes within each half); index rides
    // along in the mantissa bits for free.
    #pragma unroll
    for (int off = 1; off < 16; off <<= 1) {
        #pragma unroll
        for (int j = 0; j < 8; ++j)
            best[j] = fminf(best[j], __shfl_xor(best[j], off, 32));
    }
    // Lane 0 holds rows rowbase+0..7 (regs 0..7); lane 16 holds rowbase+8..15.
    if (m == 0) {
        #pragma unroll
        for (int j = 0; j < 8; ++j)
            idxOut[rowbase + half * 8 + j] =
                (float)(__float_as_uint(best[j]) & 1023u);
    }
}

// ---------------------------------------------------------------------------
// Kernel 3: gather quantized = codebook[idx]. One float4 per thread.
// Non-temporal store: keep the 134MB output stream from evicting L2-resident
// codebook/index data.
// ---------------------------------------------------------------------------
__global__ void vq_gather(const float* __restrict__ cb,
                          const float* __restrict__ idxF,
                          float* __restrict__ out) {
    int i = blockIdx.x * blockDim.x + threadIdx.x;   // f32x4 index, < NROWS*16
    int row  = i >> 4;
    int q    = i & 15;
    int code = (int)idxF[row];
    f32x4 v = ((const f32x4*)cb)[code * 16 + q];
    __builtin_nontemporal_store(v, (f32x4*)out + i);
}

// ---------------------------------------------------------------------------
extern "C" void kernel_launch(void* const* d_in, const int* in_sizes, int n_in,
                              void* d_out, int out_size, void* d_ws, size_t ws_size,
                              hipStream_t stream) {
    const float* x  = (const float*)d_in[0];   // [64, 8192, 64] fp32
    const float* cb = (const float*)d_in[1];   // [1024, 64] fp32

    float* out    = (float*)d_out;                         // quantized (33.5M floats)
    float* idxOut = out + (size_t)NROWS * DIM;             // indices as floats

    unsigned short* arranged = (unsigned short*)d_ws;      // 131072 bytes
    float* norms = (float*)((char*)d_ws + ARR_BYTES);      // 4096 bytes

    vq_prep<<<(TILES * 2 * 32 * 16) / 256, 256, 0, stream>>>(cb, arranged, norms);
    vq_main<<<NROWS / 128, 256, SMEM_BYTES, stream>>>(x, arranged, norms, idxOut);
    vq_gather<<<(NROWS * 16) / 256, 256, 0, stream>>>(cb, idxOut, out);
}
